// TransformerBlock_69303592288548
// MI455X (gfx1250) — matrix operand order repaired and checked by reading, not compiled, not run on hardware
//
#include <hip/hip_runtime.h>
#include <math.h>

// ---------------------------------------------------------------------------
// Transformer block for MI455X (gfx1250, wave32, WMMA + TDM).
// B=2, S=2048, D=768, H=12, HD=64, F=3072.  All GEMMs in bf16 WMMA, fp32 accum.
// ---------------------------------------------------------------------------

typedef __bf16 bf16;
typedef __attribute__((ext_vector_type(16))) __bf16 v16bf;
typedef __attribute__((ext_vector_type(8)))  __bf16 v8bf;
typedef __attribute__((ext_vector_type(8)))  float   v8f;
typedef unsigned int u32x4 __attribute__((ext_vector_type(4)));
typedef int          i32x4 __attribute__((ext_vector_type(4)));
typedef int          i32x8 __attribute__((ext_vector_type(8)));

#define B_   2
#define S_   2048
#define D_   768
#define H_   12
#define HD_  64
#define F_   3072
#define M_   (B_ * S_)    // 4096 rows of the token stream
#define N3D  (3 * D_)     // 2304

#if __has_builtin(__builtin_amdgcn_tensor_load_to_lds) && \
    __has_builtin(__builtin_amdgcn_s_wait_tensorcnt)
#define USE_TDM 1
#else
#define USE_TDM 0
#endif

union FragU { v16bf v; v8bf h[2]; };

// A-operand (16x32 bf16, M x K): lane L holds row M=L&15.
// lanes 0-15: K 0..7 in v0-3, K 16..23 in v4-7; lanes 16-31: K 8..15, 24..31.
__device__ __forceinline__ v16bf load_a_frag(const bf16* __restrict__ A, int lda) {
  const int lane = threadIdx.x & 31;
  const int row  = lane & 15;
  const int kl   = (lane >> 4) << 3;          // 0 or 8
  const bf16* p  = A + (size_t)row * lda + kl;
  FragU u;
  u.h[0] = *(const v8bf*)(p);
  u.h[1] = *(const v8bf*)(p + 16);
  return u.v;
}

// B-operand (32x16 bf16, K x N) sourced from [N,K] row-major storage:
// lane L holds column N=L&15; lanes 0-15: K 0..7, 16..23; lanes 16-31: K 8..15, 24..31.
__device__ __forceinline__ v16bf load_b_frag(const bf16* __restrict__ W, int ldw) {
  const int lane = threadIdx.x & 31;
  const int col  = lane & 15;
  const int kl   = (lane >> 4) << 3;          // 0 or 8
  const bf16* p  = W + (size_t)col * ldw + kl;
  FragU u;
  u.h[0] = *(const v8bf*)(p);
  u.h[1] = *(const v8bf*)(p + 16);
  return u.v;
}

__device__ __forceinline__ v8f wmma_bf16(v16bf a, v16bf b, v8f c) {
  return __builtin_amdgcn_wmma_f32_16x16x32_bf16(false, a, false, b,
                                                 (short)0, c, false, false);
}

__device__ __forceinline__ v8f vzero8() {
  v8f z = {0.f, 0.f, 0.f, 0.f, 0.f, 0.f, 0.f, 0.f};
  return z;
}

// ---------------------------------------------------------------------------
// 16-lane butterfly reductions with DPP16 (pure VALU; each DPP "row" of 16
// lanes is exactly one half of the wave, matching the WMMA C-layout halves).
// ---------------------------------------------------------------------------
template <int CTRL>
__device__ __forceinline__ float dpp_bf(float x) {
  return __int_as_float(
      __builtin_amdgcn_mov_dpp(__float_as_int(x), CTRL, 0xF, 0xF, true));
}
__device__ __forceinline__ float red_max16(float x) {
  x = fmaxf(x, dpp_bf<0xB1>(x));    // quad_perm [1,0,3,2]  (xor 1)
  x = fmaxf(x, dpp_bf<0x4E>(x));    // quad_perm [2,3,0,1]  (xor 2)
  x = fmaxf(x, dpp_bf<0x141>(x));   // row_half_mirror      (xor 7)
  x = fmaxf(x, dpp_bf<0x140>(x));   // row_mirror           (xor 15)
  return x;
}
__device__ __forceinline__ float red_sum16(float x) {
  x += dpp_bf<0xB1>(x);
  x += dpp_bf<0x4E>(x);
  x += dpp_bf<0x141>(x);
  x += dpp_bf<0x140>(x);
  return x;
}

#if USE_TDM
// ---------------------------------------------------------------------------
// Tensor Data Mover: copy a 2D tile (32 bf16 wide x `rows` tall, row stride
// `stride_elems`) from global into LDS.  Hand-packed D# per ISA 08 §8.3/8.4:
//   group0: count=1 | lds_addr | global_addr[56:0] | type=2
//   group1: data_size=1(2B), tensor_dim0=stride, tensor_dim1=rows,
//           tile_dim0=32, tile_dim1=rows, tensor_dim0_stride=stride
// groups 2/3 zero (2D tensor).  Tracked by TENSORcnt.
// ---------------------------------------------------------------------------
__device__ __forceinline__ void tdm_load_b_tile(const bf16* gsrc,
                                                unsigned lds_addr,
                                                int rows, int stride_elems) {
  const unsigned long long ga = (unsigned long long)(size_t)gsrc;
  u32x4 g0;
  g0[0] = 1u;                                           // count=1, user mode
  g0[1] = lds_addr;                                     // LDS byte address
  g0[2] = (unsigned)ga;                                 // global_addr[31:0]
  g0[3] = (unsigned)((ga >> 32) & 0x01FFFFFFu) | (2u << 30);  // [56:32]|type=2
  const unsigned dim0 = (unsigned)stride_elems;         // tensor_dim0 (>= tile)
  const unsigned dim1 = (unsigned)rows;                 // tensor_dim1
  i32x8 g1;
  g1[0] = (int)(1u << 16);                              // data_size = 2 bytes
  g1[1] = (int)((dim0 & 0xFFFFu) << 16);                // barrier=0 | dim0.lo
  g1[2] = (int)(((dim0 >> 16) & 0xFFFFu) | ((dim1 & 0xFFFFu) << 16));
  g1[3] = (int)(((dim1 >> 16) & 0xFFFFu) | (32u << 16));        // | tile_dim0=32
  g1[4] = (int)(dim1 & 0xFFFFu);                        // tile_dim1 | tile_dim2=0
  g1[5] = stride_elems;                                 // tensor_dim0_stride lo32
  g1[6] = 0;                                            // stride hi | dim1_stride
  g1[7] = 0;
  const i32x4 gz = {0, 0, 0, 0};
#if defined(__clang_major__) && (__clang_major__ >= 23)
  const i32x8 gz8 = {0, 0, 0, 0, 0, 0, 0, 0};
  __builtin_amdgcn_tensor_load_to_lds(g0, g1, gz, gz, gz8, 0);
#else
  __builtin_amdgcn_tensor_load_to_lds(g0, g1, gz, gz, 0);
#endif
}
#endif  // USE_TDM

// ---------------------------------------------------------------------------
// fp32 -> bf16 conversion (weights, once per launch; all stays in L2)
// ---------------------------------------------------------------------------
__global__ void f32_to_bf16_kernel(const float* __restrict__ in,
                                   bf16* __restrict__ out, int n) {
  int i = blockIdx.x * 256 + threadIdx.x;
  if (i < n) out[i] = (bf16)in[i];
}

// ---------------------------------------------------------------------------
// Row LayerNorm (D=768), fp32 in -> bf16 out (GEMM feed). One row per block.
// ---------------------------------------------------------------------------
__global__ __launch_bounds__(256) void layernorm_bf16_kernel(
    const float* __restrict__ x, const float* __restrict__ w,
    const float* __restrict__ bvec, bf16* __restrict__ y) {
  __shared__ float reds[8], reds2[8];
  const int row = blockIdx.x;
  const float* xr = x + (size_t)row * D_;
  float s = 0.f, s2 = 0.f;
  for (int i = threadIdx.x; i < D_; i += 256) {
    float v = xr[i]; s += v; s2 += v * v;
  }
  for (int m = 1; m < 32; m <<= 1) { s += __shfl_xor(s, m); s2 += __shfl_xor(s2, m); }
  const int wave = threadIdx.x >> 5;
  const int lane = threadIdx.x & 31;
  if (lane == 0) { reds[wave] = s; reds2[wave] = s2; }
  __syncthreads();
  if (wave == 0) {
    s  = (lane < 8) ? reds[lane]  : 0.f;
    s2 = (lane < 8) ? reds2[lane] : 0.f;
    for (int m = 1; m < 8; m <<= 1) { s += __shfl_xor(s, m); s2 += __shfl_xor(s2, m); }
    if (lane == 0) { reds[0] = s; reds2[0] = s2; }
  }
  __syncthreads();
  s = reds[0]; s2 = reds2[0];
  const float mean = s * (1.0f / D_);
  const float var  = s2 * (1.0f / D_) - mean * mean;
  const float rstd = rsqrtf(var + 1e-5f);
  for (int i = threadIdx.x; i < D_; i += 256) {
    y[(size_t)row * D_ + i] = (bf16)((xr[i] - mean) * rstd * w[i] + bvec[i]);
  }
}

// ---------------------------------------------------------------------------
// Tiled WMMA GEMM: C[M,N] = A[M,K] @ W[N,K]^T (+ epilogue).
// Block = 8 waves (4 along M x 2 along N). Wave tile = (MI*16) x (NI*16).
// The B (weight) tile -- shared by all 4 M-waves -- is staged into LDS by the
// Tensor Data Mover, double-buffered so the copy overlaps the WMMAs; B-frag
// reads become ds_load_b128.  A frags come straight from global (L2-resident).
// EPI: 0 bias->bf16 | 1 bias+residual->f32 | 2 bias+GELU->bf16
//      3 QKV: bias->bf16 qkv buffer, V part also transposed into vt[b,h,d,s]
// ---------------------------------------------------------------------------
template <int EPI, int MI, int NI>
__global__ __launch_bounds__(256) void gemm_wmma_kernel(
    const bf16* __restrict__ A, const bf16* __restrict__ W,
    const float* __restrict__ bias, const float* __restrict__ resid,
    float* __restrict__ outF, bf16* __restrict__ outH, bf16* __restrict__ vt,
    int M, int N, int K) {
  const int wave = threadIdx.x >> 5;
  const int lane = threadIdx.x & 31;
  const int mw = wave >> 1;                       // 0..3
  const int nw = wave & 1;                        // 0..1
  const int row0 = blockIdx.y * (64 * MI) + mw * (MI * 16);
  const int col0 = blockIdx.x * (32 * NI) + nw * (NI * 16);

  v8f acc[MI][NI];
#pragma unroll
  for (int i = 0; i < MI; ++i)
#pragma unroll
    for (int j = 0; j < NI; ++j) acc[i][j] = vzero8();

  const bf16* Arow0 = A + (size_t)row0 * K;

#if USE_TDM
  constexpr int BROWS = NI * 32;                  // B rows per block (=128)
  __shared__ __align__(16) bf16 btile[2][BROWS][32];
  const bf16* Wblk = W + (size_t)blockIdx.x * BROWS * K;
  const int nk = K >> 5;
  if (wave == 0)
    tdm_load_b_tile(Wblk, (unsigned)(size_t)&btile[0][0][0], BROWS, K);
  for (int kt = 0; kt < nk; ++kt) {
    const int k0 = kt << 5;
    if (wave == 0) __builtin_amdgcn_s_wait_tensorcnt(0);  // tile kt arrived
    __syncthreads();                                      // publish to block
    if (wave == 0 && kt + 1 < nk)                         // prefetch kt+1
      tdm_load_b_tile(Wblk + (size_t)(k0 + 32),
                      (unsigned)(size_t)&btile[(kt + 1) & 1][0][0], BROWS, K);
    __builtin_prefetch(Arow0 + k0 + 256, 0, 1);
    v16bf af[MI];
    v16bf bfr[NI];
#pragma unroll
    for (int i = 0; i < MI; ++i)
      af[i] = load_a_frag(Arow0 + (size_t)(i * 16) * K + k0, K);
    const bf16* bb = &btile[kt & 1][nw * (NI * 16)][0];
#pragma unroll
    for (int j = 0; j < NI; ++j)
      bfr[j] = load_b_frag(bb + (size_t)(j * 16) * 32, 32);
#pragma unroll
    for (int i = 0; i < MI; ++i)
#pragma unroll
      for (int j = 0; j < NI; ++j)
        acc[i][j] = wmma_bf16(af[i], bfr[j], acc[i][j]);
    // no trailing barrier needed: next overwrite of btile[(kt+1)&1] is issued
    // only after the *next* top-of-loop barrier, and __syncthreads waits DScnt.
  }
#else
  const bf16* Wrow0 = W + (size_t)col0 * K;
  for (int k0 = 0; k0 < K; k0 += 32) {
    __builtin_prefetch(Arow0 + k0 + 256, 0, 1);
    v16bf af[MI];
    v16bf bfr[NI];
#pragma unroll
    for (int i = 0; i < MI; ++i)
      af[i] = load_a_frag(Arow0 + (size_t)(i * 16) * K + k0, K);
#pragma unroll
    for (int j = 0; j < NI; ++j)
      bfr[j] = load_b_frag(Wrow0 + (size_t)(j * 16) * K + k0, K);
#pragma unroll
    for (int i = 0; i < MI; ++i)
#pragma unroll
      for (int j = 0; j < NI; ++j)
        acc[i][j] = wmma_bf16(af[i], bfr[j], acc[i][j]);
  }
#endif

  const int hi = lane >> 4;
  const int nn = lane & 15;
#pragma unroll
  for (int i = 0; i < MI; ++i) {
#pragma unroll
    for (int j = 0; j < NI; ++j) {
#pragma unroll
      for (int r = 0; r < 8; ++r) {
        const int m = row0 + i * 16 + r + hi * 8;   // C layout: VGPR r = row r / r+8
        const int n = col0 + j * 16 + nn;           // lane n within 16
        float v = acc[i][j][r] + bias[n];
        if (EPI == 0) {
          outH[(size_t)m * N + n] = (bf16)v;
        } else if (EPI == 1) {
          outF[(size_t)m * N + n] = v + resid[(size_t)m * N + n];
        } else if (EPI == 2) {
          const float g = 0.5f * v * (1.0f + erff(v * 0.70710678f)); // exact GELU
          outH[(size_t)m * N + n] = (bf16)g;
        } else {  // EPI == 3 : QKV projection
          outH[(size_t)m * N + n] = (bf16)v;
          if (n >= 2 * D_) {                        // V slice -> transposed [b,h,d,s]
            const int s      = m & (S_ - 1);
            const int bb2    = m >> 11;
            const int within = n - 2 * D_;
            const int h      = within >> 6;
            const int d      = within & 63;
            vt[(((size_t)(bb2 * H_ + h)) * HD_ + d) * S_ + s] = (bf16)v;
          }
        }
      }
    }
  }
}

// ---------------------------------------------------------------------------
// Flash attention: one wave per 16-query tile per (b,h). Streams 32-key tiles:
// QK^T via WMMA, online softmax with DPP16 butterfly reductions (VALU-only),
// P goes through LDS (C-layout -> A-layout) and PV via WMMA into 16x64 fp32.
// Scores kept in log2 domain: SCL = log2(e)/sqrt(HD), probs = exp2().
// ---------------------------------------------------------------------------
__global__ __launch_bounds__(256) void attn_kernel(
    const bf16* __restrict__ qkv, const bf16* __restrict__ vt,
    bf16* __restrict__ obuf) {
  __shared__ __align__(16) bf16 plds[8][16 * 32];   // 1 KB per wave
  const int wave = threadIdx.x >> 5;
  const int lane = threadIdx.x & 31;
  const int t  = blockIdx.x * 8 + wave;             // 0..3071
  const int qt = t & 127;                           // S/16 = 128 q-tiles
  const int h  = (t >> 7) % H_;
  const int b  = t / (128 * H_);
  const int q0 = qt * 16;
  const int hi = lane >> 4;
  const int nn = lane & 15;
  const size_t rowbase = (size_t)b * S_;
  const float SCL = 0.18033688011112042f;           // log2(e) / sqrt(64)

  const bf16* Qp = qkv + (rowbase + q0) * N3D + h * HD_;
  const v16bf qa0 = load_a_frag(Qp, N3D);
  const v16bf qa1 = load_a_frag(Qp + 32, N3D);

  v8f o[4];
#pragma unroll
  for (int j = 0; j < 4; ++j) o[j] = vzero8();
  float mrun[8], lrun[8];
#pragma unroll
  for (int r = 0; r < 8; ++r) { mrun[r] = -INFINITY; lrun[r] = 0.f; }

  const int nkt = (q0 + 16 + 31) >> 5;              // 32-wide key tiles, causal span
  for (int kt = 0; kt < nkt; ++kt) {
    const int n0 = kt * 32;
    const bf16* Kp0 = qkv + (rowbase + n0) * N3D + D_ + h * HD_;
    const bf16* Kp1 = Kp0 + (size_t)16 * N3D;
    v8f c0 = vzero8(), c1 = vzero8();
    c0 = wmma_bf16(qa0, load_b_frag(Kp0, N3D), c0);
    c0 = wmma_bf16(qa1, load_b_frag(Kp0 + 32, N3D), c0);
    c1 = wmma_bf16(qa0, load_b_frag(Kp1, N3D), c1);
    c1 = wmma_bf16(qa1, load_b_frag(Kp1 + 32, N3D), c1);

#pragma unroll
    for (int r = 0; r < 8; ++r) {
      const int qpos = q0 + r + hi * 8;
      const int kp0 = n0 + nn;
      const int kp1 = n0 + 16 + nn;
      float x0 = (kp0 <= qpos) ? c0[r] * SCL : -INFINITY;
      float x1 = (kp1 <= qpos) ? c1[r] * SCL : -INFINITY;
      const float rm = red_max16(fmaxf(x0, x1));
      const float mnew = fmaxf(mrun[r], rm);
      const float corr = exp2f(mrun[r] - mnew);
      const float p0 = exp2f(x0 - mnew);
      const float p1 = exp2f(x1 - mnew);
      const float rs = red_sum16(p0 + p1);
      lrun[r] = lrun[r] * corr + rs;
      mrun[r] = mnew;
#pragma unroll
      for (int j = 0; j < 4; ++j) o[j][r] *= corr;
      const int mloc = r + hi * 8;
      plds[wave][mloc * 32 + nn]      = (bf16)p0;
      plds[wave][mloc * 32 + 16 + nn] = (bf16)p1;
    }
    asm volatile("s_wait_dscnt 0" ::: "memory");    // P stores visible before reload

    const v16bf pa = load_a_frag(&plds[wave][0], 32);
    const bf16* Vp = vt + ((size_t)(b * H_ + h) * HD_) * S_ + n0;
#pragma unroll
    for (int j = 0; j < 4; ++j) {
      v16bf vb = load_b_frag(Vp + (size_t)(j * 16) * S_, S_);
      o[j] = wmma_bf16(pa, vb, o[j]);
    }
  }

#pragma unroll
  for (int r = 0; r < 8; ++r) {
    const float inv = __builtin_amdgcn_rcpf(lrun[r]);
    const size_t m = rowbase + q0 + r + hi * 8;
#pragma unroll
    for (int j = 0; j < 4; ++j) {
      obuf[m * D_ + h * HD_ + j * 16 + nn] = (bf16)(o[j][r] * inv);
    }
  }
}

// ---------------------------------------------------------------------------
// Host-side orchestration
// ---------------------------------------------------------------------------
extern "C" void kernel_launch(void* const* d_in, const int* in_sizes, int n_in,
                              void* d_out, int out_size, void* d_ws, size_t ws_size,
                              hipStream_t stream) {
  (void)in_sizes; (void)n_in; (void)out_size; (void)ws_size;
  const float* x     = (const float*)d_in[0];
  /* d_in[1] = causal mask, computed analytically -> unused */
  const float* qkv_w = (const float*)d_in[2];
  const float* qkv_b = (const float*)d_in[3];
  const float* out_w = (const float*)d_in[4];
  const float* out_b = (const float*)d_in[5];
  const float* fc1_w = (const float*)d_in[6];
  const float* fc1_b = (const float*)d_in[7];
  const float* fc2_w = (const float*)d_in[8];
  const float* fc2_b = (const float*)d_in[9];
  const float* ln1_w = (const float*)d_in[10];
  const float* ln1_b = (const float*)d_in[11];
  const float* ln2_w = (const float*)d_in[12];
  const float* ln2_b = (const float*)d_in[13];
  float* out = (float*)d_out;

  // ~96 MB of scratch (fits in the 192 MB L2 alongside weights)
  char* ws = (char*)d_ws;
  auto take = [&](size_t bytes) -> char* {
    char* p = ws; ws += (bytes + 255) & ~(size_t)255; return p;
  };
  bf16*  qkvw_h = (bf16*)take((size_t)N3D * D_ * 2);
  bf16*  outw_h = (bf16*)take((size_t)D_ * D_ * 2);
  bf16*  fc1w_h = (bf16*)take((size_t)F_ * D_ * 2);
  bf16*  fc2w_h = (bf16*)take((size_t)D_ * F_ * 2);
  bf16*  hx     = (bf16*)take((size_t)M_ * D_ * 2);
  bf16*  qkvb   = (bf16*)take((size_t)M_ * N3D * 2);
  bf16*  vt     = (bf16*)take((size_t)B_ * H_ * HD_ * S_ * 2);
  bf16*  obuf   = (bf16*)take((size_t)M_ * D_ * 2);
  float* x1     = (float*)take((size_t)M_ * D_ * 4);
  bf16*  h2     = (bf16*)take((size_t)M_ * D_ * 2);
  bf16*  ff1    = (bf16*)take((size_t)M_ * F_ * 2);

  auto cvt = [&](const float* a, bf16* o, int n) {
    f32_to_bf16_kernel<<<(n + 255) / 256, 256, 0, stream>>>(a, o, n);
  };
  cvt(qkv_w, qkvw_h, N3D * D_);
  cvt(out_w, outw_h, D_ * D_);
  cvt(fc1_w, fc1w_h, F_ * D_);
  cvt(fc2_w, fc2w_h, D_ * F_);

  // 1. LN1 -> bf16
  layernorm_bf16_kernel<<<M_, 256, 0, stream>>>(x, ln1_w, ln1_b, hx);
  // 2. QKV projection (+V transpose epilogue); 256x128 blocks, 64x64 waves
  gemm_wmma_kernel<3, 4, 4><<<dim3(N3D / 128, M_ / 256), 256, 0, stream>>>(
      hx, qkvw_h, qkv_b, nullptr, nullptr, qkvb, vt, M_, N3D, D_);
  // 3. causal flash attention (3072 wave-tiles, 8 per block)
  attn_kernel<<<(B_ * H_ * (S_ / 16)) / 8, 256, 0, stream>>>(qkvb, vt, obuf);
  // 4. out-proj + bias + residual -> fp32 residual stream x1 (128x128 blocks)
  gemm_wmma_kernel<1, 2, 4><<<dim3(D_ / 128, M_ / 128), 256, 0, stream>>>(
      obuf, outw_h, out_b, x, x1, nullptr, nullptr, M_, D_, D_);
  // 5. LN2 -> bf16
  layernorm_bf16_kernel<<<M_, 256, 0, stream>>>(x1, ln2_w, ln2_b, h2);
  // 6. FC1 + bias + exact GELU -> bf16; 256x128 blocks
  gemm_wmma_kernel<2, 4, 4><<<dim3(F_ / 128, M_ / 256), 256, 0, stream>>>(
      h2, fc1w_h, fc1_b, nullptr, nullptr, ff1, nullptr, M_, F_, D_);
  // 7. FC2 + bias + residual -> fp32 output; 128x128 blocks
  gemm_wmma_kernel<1, 2, 4><<<dim3(D_ / 128, M_ / 128), 256, 0, stream>>>(
      ff1, fc2w_h, fc2_b, x1, out, nullptr, nullptr, M_, D_, F_);
}
